// RecurrentBlock_46978352284316
// MI455X (gfx1250) — matrix operand order repaired and checked by reading, not compiled, not run on hardware
//
#include <hip/hip_runtime.h>
#include <hip/hip_bf16.h>

typedef __bf16 bf16;
typedef __attribute__((ext_vector_type(8)))  float v8f;
typedef __attribute__((ext_vector_type(8)))  bf16  v8bf;
typedef __attribute__((ext_vector_type(16))) bf16  v16bf;

#define B_   2
#define H_   256
#define W_   512
#define PH_  258
#define PW_  514
#define MAXD_ 4
#define ND_  9

// padded NHWC strides (elements)
#define ROW32 (PW_*32)
#define IMG32 ((long)PH_*ROW32)
#define XCS   384
#define ROWX  (PW_*XCS)
#define IMGX  ((long)PH_*ROWX)
#define HCS   48
#define ROWH  (PW_*HCS)
#define IMGH  ((long)PH_*ROWH)

// channel bases inside X: [g3 | g2 | g1 | g0 | cost | a0]
#define CH_G3   0
#define CH_G2   32
#define CH_G1   96
#define CH_G0   192
#define CH_COST 320
#define CH_A0   329

#define STATP 64

__device__ __forceinline__ v8f wmma_bf16(v16bf a, v16bf b, v8f c) {
  // (neg_a, A, neg_b, B, c_mod, C, reuse_a, reuse_b)
  return __builtin_amdgcn_wmma_f32_16x16x32_bf16(false, a, false, b, (short)0, c, false, false);
}

// -------------------- generic 3x3 SAME conv as implicit GEMM --------------------
// Block: 128 threads (4 waves), 64-pixel row tile. Each wave: 64 pixels x 16 couts
// -> 4 v_wmma_f32_16x16x32_bf16 per K-step sharing one B fragment.
// A tiles are staged memory->LDS with CDNA5 async copies
// (global_load_async_to_lds_b128, ASYNCcnt), 4-buffer pipeline, prefetch dist 2.
// in/out/res point at the PADDING CORNER of NHWC-padded buffers (channel base
// pre-added). Interior pixel (y,x) lives at row y+1, col x+1.
// Weights wp packed as [(nt*9+t)*cChunks+j][n(16)][k(32)] bf16 (zero padded) --
// linear in the K-step index, so the B pointer advances by 512 elems/step.
__global__ __launch_bounds__(128)
void conv3x3_wmma(const bf16* __restrict__ in, int inCS, long inRow, long inImg, int Cin,
                  const bf16* __restrict__ wp, const float* __restrict__ bias,
                  bf16* __restrict__ out, int outCS, long outRow, long outImg, int Cout,
                  int mode,  // 0 none, 1 relu, 2 lrelu, 3 residual-add + relu
                  const bf16* __restrict__ res, int resCS, long resRow, long resImg)
{
  const int xt    = blockIdx.x * 64;
  const int y     = blockIdx.y;
  const int b     = blockIdx.z;
  const int tid   = threadIdx.x;
  const int waveS = __builtin_amdgcn_readfirstlane(tid >> 5);  // scalar wave id
  const int lane  = tid & 31;

  __shared__ bf16 aStage[4][64][32];   // 16 KB, 4-deep async pipeline

  const int cChunks = (Cin + 31) >> 5;
  const int S       = 9 * cChunks;
  const int nTiles  = (Cout + 15) >> 4;

  const bf16* inB = in + (long)b * inImg;

  const int sm  = tid >> 1;          // staging pixel 0..63
  const int spp = (tid & 1) * 16;    // staging channel half (16 bf16 = 32B)

  // incremental staging-pointer deltas (uniform)
  const long dCol = (long)inCS - (long)(cChunks - 1) * 32;                 // next kx
  const long dRow = inRow - 2 * (long)inCS - (long)(cChunks - 1) * 32;     // next ky

  const int n    = lane & 15;
  const int koff = (lane & 16) ? 16 : 0;  // A frag K half per lane half
  const int kh   = (lane & 16) ? 16 : 0;  // B frag K half per lane half
  const int aOff = n * 32 + koff;

  // 32-bit LDS byte address of this thread's staging slot in buffer 0
  // (generic LDS pointer low 32 bits == LDS address)
  const unsigned ldsBase0 = (unsigned)(size_t)&aStage[0][sm][spp];

  for (int ntBase = 0; ntBase < nTiles; ntBase += 4) {
    const int  nt  = ntBase + waveS;
    const bool act = (nt < nTiles);       // scalar -> s_cbranch, EXEC stays all-1s

    v8f z = {0.f,0.f,0.f,0.f,0.f,0.f,0.f,0.f};
    v8f acc[4] = {z, z, z, z};

    const bf16* wpP  = wp + (long)nt * S * 512 + n * 32 + kh;  // +512 per step
    const bf16* sPtr = inB + (long)y * inRow + (long)(xt + sm) * inCS + spp;
    int jS = 0, txS = 0;

    auto advance = [&]() {               // branchless uniform control
      bool wrap = (jS == cChunks - 1);
      long d    = wrap ? ((txS == 2) ? dRow : dCol) : (long)32;
      txS       = wrap ? ((txS == 2) ? 0 : txS + 1) : txS;
      jS        = wrap ? 0 : jS + 1;
      sPtr += d;
    };
    auto asyncStage = [&](int bufI) {    // 32B memory->LDS, no VGPR round-trip
      unsigned lds32 = ldsBase0 + (unsigned)bufI * (64u * 32u * 2u);
      asm volatile("global_load_async_to_lds_b128 %0, %1, off"
                   :: "v"(lds32), "v"(sPtr) : "memory");
      asm volatile("global_load_async_to_lds_b128 %0, %1, off offset:16"
                   :: "v"(lds32), "v"(sPtr) : "memory");
      advance();
    };

    // prologue: prefetch steps 0 and 1 (S >= 9 always)
    asyncStage(0);
    asyncStage(1);

    for (int s = 0; s < S; ++s) {
      const int buf = s & 3;
      if (s + 2 < S) {
        asyncStage((s + 2) & 3);
        // 3 stages (s, s+1, s+2) may be outstanding = 6 ops; <=4 completes stage s
        asm volatile("s_wait_asynccnt 0x4" ::: "memory");
      } else if (s + 1 < S) {
        asm volatile("s_wait_asynccnt 0x2" ::: "memory");
      } else {
        asm volatile("s_wait_asynccnt 0x0" ::: "memory");
      }
      __syncthreads();                   // everyone's stage(s) now visible in LDS
      if (act) {
        const bf16* ab = &aStage[buf][0][0];
        v16bf bfrag = *(const v16bf*)wpP;        // one 32B load, 4 WMMAs
        #pragma unroll
        for (int q = 0; q < 4; ++q) {
          v8bf lo = *(const v8bf*)(ab + q * 512 + aOff);
          v8bf hi = *(const v8bf*)(ab + q * 512 + aOff + 8);
          v16bf af = __builtin_shufflevector(lo, hi,
                      0,1,2,3,4,5,6,7,8,9,10,11,12,13,14,15);
          acc[q] = wmma_bf16(af, bfrag, acc[q]);
        }
      }
      wpP += 512;
    }
    __syncthreads();                     // protect buffers before next pass restages

    if (act) {
      const int cout = nt * 16 + n;
      if (cout < Cout) {
        const float bi   = bias[cout];
        const bool  hiH  = (lane & 16) != 0;
        bf16* outB       = out + (long)b * outImg + (long)(y + 1) * outRow;
        const bf16* resB = res + (long)b * resImg + (long)(y + 1) * resRow;
        #pragma unroll
        for (int q = 0; q < 4; ++q) {
          #pragma unroll
          for (int r = 0; r < 8; ++r) {
            int px = xt + q * 16 + (hiH ? 8 + r : r);
            float v = acc[q][r] + bi;
            if (mode == 3) v += (float)resB[(long)(px + 1) * resCS + cout];
            if (mode == 1 || mode == 3) v = fmaxf(v, 0.f);
            else if (mode == 2)         v = (v > 0.f) ? v : 0.1f * v;
            outB[(long)(px + 1) * outCS + cout] = (bf16)v;
          }
        }
      }
    }
  }
}

// -------------------- weight packing (OIHW f32 -> WMMA B layout bf16) ----------
__global__ void pack_w(const float* __restrict__ w, bf16* __restrict__ wp,
                       int Cin, int Cout, int cChunks, long total)
{
  long i = (long)blockIdx.x * 256 + threadIdx.x;
  if (i >= total) return;
  int k = (int)(i & 31);
  int n = (int)((i >> 5) & 15);
  long rest = i >> 9;
  int j  = (int)(rest % cChunks); rest /= cChunks;
  int t  = (int)(rest % 9);
  int nt = (int)(rest / 9);
  int cout = nt * 16 + n;
  int cin  = (j << 5) + k;
  float v = 0.f;
  if (cout < Cout && cin < Cin) v = w[((long)cout * Cin + cin) * 9 + t];
  wp[i] = (bf16)v;
}

// -------------------- deterministic BN statistics ------------------------------
__global__ __launch_bounds__(256)
void stats_nchw(const float* __restrict__ x, int C, long HW, long N, float* __restrict__ part)
{
  int c = blockIdx.x, p = blockIdx.y;
  long chunk = (N + STATP - 1) / STATP;
  long s0 = (long)p * chunk, s1 = s0 + chunk; if (s1 > N) s1 = N;
  float su = 0.f, sq = 0.f;
  for (long i = s0 + threadIdx.x; i < s1; i += 256) {
    long b = i / HW, r = i - b * HW;
    float v = x[((long)b * C + c) * HW + r];
    su += v; sq += v * v;
  }
  __shared__ float sh[256][2];
  sh[threadIdx.x][0] = su; sh[threadIdx.x][1] = sq;
  __syncthreads();
  for (int st = 128; st > 0; st >>= 1) {
    if (threadIdx.x < st) {
      sh[threadIdx.x][0] += sh[threadIdx.x + st][0];
      sh[threadIdx.x][1] += sh[threadIdx.x + st][1];
    }
    __syncthreads();
  }
  if (threadIdx.x == 0) {
    part[((long)c * STATP + p) * 2]     = sh[0][0];
    part[((long)c * STATP + p) * 2 + 1] = sh[0][1];
  }
}

__global__ __launch_bounds__(256)
void stats_nhwc(const bf16* __restrict__ baseI, int CS, long row, long img,
                float* __restrict__ part)
{
  int c = blockIdx.x, p = blockIdx.y;
  const long HW = (long)H_ * W_, N = (long)B_ * HW;
  long chunk = (N + STATP - 1) / STATP;
  long s0 = (long)p * chunk, s1 = s0 + chunk; if (s1 > N) s1 = N;
  float su = 0.f, sq = 0.f;
  for (long i = s0 + threadIdx.x; i < s1; i += 256) {
    long b = i / HW, r = i - b * HW;
    int y = (int)(r / W_), x = (int)(r - (long)y * W_);
    float v = (float)baseI[b * img + (long)y * row + (long)x * CS + c];
    su += v; sq += v * v;
  }
  __shared__ float sh[256][2];
  sh[threadIdx.x][0] = su; sh[threadIdx.x][1] = sq;
  __syncthreads();
  for (int st = 128; st > 0; st >>= 1) {
    if (threadIdx.x < st) {
      sh[threadIdx.x][0] += sh[threadIdx.x + st][0];
      sh[threadIdx.x][1] += sh[threadIdx.x + st][1];
    }
    __syncthreads();
  }
  if (threadIdx.x == 0) {
    part[((long)c * STATP + p) * 2]     = sh[0][0];
    part[((long)c * STATP + p) * 2 + 1] = sh[0][1];
  }
}

__global__ void stats_fin(const float* __restrict__ part, int C, float invN,
                          const float* __restrict__ g, const float* __restrict__ bta,
                          float* __restrict__ ss)
{
  int c = threadIdx.x;
  if (c >= C) return;
  float su = 0.f, sq = 0.f;
  for (int p = 0; p < STATP; ++p) {
    su += part[((long)c * STATP + p) * 2];
    sq += part[((long)c * STATP + p) * 2 + 1];
  }
  float m = su * invN, v = sq * invN - m * m;
  float inv = rsqrtf(v + 1e-5f);
  float sc  = g[c] * inv;
  ss[c * 2]     = sc;
  ss[c * 2 + 1] = bta[c] - m * sc;
}

// -------------------- BN apply / layout conversion -----------------------------
__global__ void bn_nchw2nhwc(const float* __restrict__ x, const float* __restrict__ ss,
                             bf16* __restrict__ corner, long total)
{
  long i = (long)blockIdx.x * 256 + threadIdx.x;
  if (i >= total) return;
  int c = (int)(i & 31);
  long p = i >> 5;
  int xx = (int)(p % W_); long q = p / W_;
  int yy = (int)(q % H_); int bb = (int)(q / H_);
  float v = x[(((long)bb * 32 + c) * H_ + yy) * W_ + xx];
  v = v * ss[c * 2] + ss[c * 2 + 1];
  corner[(long)bb * IMG32 + (long)(yy + 1) * ROW32 + (long)(xx + 1) * 32 + c] = (bf16)v;
}

__global__ void bn_nhwc(const bf16* __restrict__ inI, int inCS, long inRow, long inImg,
                        bf16* __restrict__ outI, int outCS, long outRow, long outImg,
                        const float* __restrict__ ss, int C, long total)
{
  long i = (long)blockIdx.x * 256 + threadIdx.x;
  if (i >= total) return;
  int c = (int)(i % C); long p = i / C;
  int xx = (int)(p % W_); long q = p / W_;
  int yy = (int)(q % H_); int bb = (int)(q / H_);
  float v = (float)inI[(long)bb * inImg + (long)yy * inRow + (long)xx * inCS + c];
  v = v * ss[c * 2] + ss[c * 2 + 1];
  outI[(long)bb * outImg + (long)yy * outRow + (long)xx * outCS + c] = (bf16)v;
}

// -------------------- warp / correlation / head --------------------------------
__global__ void warp_k(const bf16* __restrict__ a1I, const float* __restrict__ disp,
                       bf16* __restrict__ outI, long total)
{
  long i = (long)blockIdx.x * 256 + threadIdx.x;
  if (i >= total) return;
  int xx = (int)(i % W_); long q = i / W_;
  int yy = (int)(q % H_); int bb = (int)(q / H_);
  float d  = disp[((long)bb * H_ + yy) * W_ + xx];
  float xs = (float)xx - d;
  float x0f = floorf(xs);
  float fr  = xs - x0f;
  int x0 = (int)x0f, x1 = x0 + 1;
  float v0 = (x0 >= 0 && x0 <= W_ - 1) ? 1.f : 0.f;
  float v1 = (x1 >= 0 && x1 <= W_ - 1) ? 1.f : 0.f;
  int c0 = x0 < 0 ? 0 : (x0 > W_ - 1 ? W_ - 1 : x0);
  int c1 = x1 < 0 ? 0 : (x1 > W_ - 1 ? W_ - 1 : x1);
  float m = (1.f - fr) * v0 + fr * v1;
  m = (m < 0.9999f) ? 0.f : 1.f;
  float w0 = (1.f - fr) * v0 * m, w1 = fr * v1 * m;
  const bf16* src = a1I + (long)bb * IMG32 + (long)yy * ROW32;
  bf16*       dst = outI + (long)bb * IMG32 + (long)yy * ROW32 + (long)xx * 32;
  #pragma unroll 4
  for (int c = 0; c < 32; ++c)
    dst[c] = (bf16)((float)src[(long)c0 * 32 + c] * w0 + (float)src[(long)c1 * 32 + c] * w1);
}

__global__ void corr_raw_k(const bf16* __restrict__ a0I, const bf16* __restrict__ a1wI,
                           float* __restrict__ cr, long total)
{
  long i = (long)blockIdx.x * 256 + threadIdx.x;
  if (i >= total) return;
  int xx = (int)(i % W_); long q = i / W_;
  int yy = (int)(q % H_); int bb = (int)(q / H_);
  const bf16* p0 = a0I + (long)bb * IMGX + (long)yy * ROWX + (long)xx * XCS;
  const bf16* p1r = a1wI + (long)bb * IMG32 + (long)yy * ROW32;
  for (int d = 0; d < ND_; ++d) {
    int x2 = xx + d - MAXD_;
    float s = 0.f;
    if (x2 >= 0 && x2 < W_) {
      const bf16* p1 = p1r + (long)x2 * 32;
      #pragma unroll 8
      for (int c = 0; c < 32; ++c) s += (float)p0[c] * (float)p1[c];
    }
    cr[((((long)bb * ND_ + d) * H_) + yy) * W_ + xx] = s * (1.f / 32.f);
  }
}

__global__ void corr_box_ca_k(const float* __restrict__ cr, const float* __restrict__ wca,
                              const float* __restrict__ bca, bf16* __restrict__ costI, long total)
{
  long i = (long)blockIdx.x * 256 + threadIdx.x;
  if (i >= total) return;
  int xx = (int)(i % W_); long q = i / W_;
  int yy = (int)(q % H_); int bb = (int)(q / H_);
  float v[ND_];
  for (int d = 0; d < ND_; ++d) {
    float s = 0.f;
    for (int dy = -1; dy <= 1; ++dy) {
      int y2 = yy + dy; if (y2 < 0 || y2 >= H_) continue;
      const float* row = cr + ((((long)bb * ND_ + d) * H_) + y2) * W_;
      for (int dx = -1; dx <= 1; ++dx) {
        int x2 = xx + dx; if (x2 < 0 || x2 >= W_) continue;
        s += row[x2];
      }
    }
    v[d] = s * (1.f / 9.f);
  }
  bf16* dst = costI + (long)bb * IMGX + (long)yy * ROWX + (long)xx * XCS;
  for (int o = 0; o < ND_; ++o) {
    float a = bca[o];
    for (int d = 0; d < ND_; ++d) a += wca[o * ND_ + d] * v[d];
    dst[o] = (bf16)a;
  }
}

__global__ void head_k(const bf16* __restrict__ hCorner, const float* __restrict__ wd2,
                       const float* __restrict__ bd2, const float* __restrict__ disp0,
                       float* __restrict__ dout, long total)
{
  long i = (long)blockIdx.x * 256 + threadIdx.x;
  if (i >= total) return;
  int xx = (int)(i % W_); long q = i / W_;
  int yy = (int)(q % H_); int bb = (int)(q / H_);
  float a = bd2[0];
  for (int t = 0; t < 9; ++t) {
    int ky = t / 3, kx = t - ky * 3;
    const bf16* p = hCorner + (long)bb * IMGH + (long)(yy + ky) * ROWH + (long)(xx + kx) * HCS;
    #pragma unroll 8
    for (int c = 0; c < 41; ++c) a += (float)p[c] * wd2[c * 9 + t];
  }
  long idx = ((long)bb * H_ + yy) * W_ + xx;
  dout[idx] = tanhf(a) * (float)MAXD_ + disp0[idx];
}

__global__ void emit_x_k(const bf16* __restrict__ xCorner, float* __restrict__ xo, long total)
{
  long i = (long)blockIdx.x * 256 + threadIdx.x;
  if (i >= total) return;
  int xx = (int)(i % W_); long q = i / W_;
  int yy = (int)(q % H_); q /= H_;
  int c  = (int)(q % 361); int bb = (int)(q / 361);
  xo[i] = (float)xCorner[(long)bb * IMGX + (long)(yy + 1) * ROWX + (long)(xx + 1) * XCS + c];
}

__global__ void zero_k(float4* __restrict__ p, long n)
{
  long i = (long)blockIdx.x * 256 + threadIdx.x;
  long stride = (long)gridDim.x * 256;
  for (; i < n; i += stride) p[i] = make_float4(0.f, 0.f, 0.f, 0.f);
}

// ==============================================================================
extern "C" void kernel_launch(void* const* d_in, const int* in_sizes, int n_in,
                              void* d_out, int out_size, void* d_ws, size_t ws_size,
                              hipStream_t stream)
{
  (void)in_sizes; (void)n_in; (void)out_size; (void)ws_size;
  const float* fimg[2] = {(const float*)d_in[0], (const float*)d_in[1]};
  const float* disp0 = (const float*)d_in[2];
  const float* bn0_g = (const float*)d_in[3];
  const float* bn0_b = (const float*)d_in[4];
  const float* w_c0  = (const float*)d_in[5];
  const float* b_c0  = (const float*)d_in[6];
  const float* w_r1  = (const float*)d_in[7];
  const float* b_r1  = (const float*)d_in[8];
  const float* w_r2  = (const float*)d_in[9];
  const float* b_r2  = (const float*)d_in[10];
  const float* bn1_g = (const float*)d_in[11];
  const float* bn1_b = (const float*)d_in[12];
  const float* w_ca  = (const float*)d_in[13];
  const float* b_ca  = (const float*)d_in[14];
  const float* w_g[4] = {(const float*)d_in[15], (const float*)d_in[17],
                         (const float*)d_in[19], (const float*)d_in[21]};
  const float* b_g[4] = {(const float*)d_in[16], (const float*)d_in[18],
                         (const float*)d_in[20], (const float*)d_in[22]};
  const float* w_d1  = (const float*)d_in[23];
  const float* b_d1  = (const float*)d_in[24];
  const float* bnd_g = (const float*)d_in[25];
  const float* bnd_b = (const float*)d_in[26];
  const float* w_d2  = (const float*)d_in[27];
  const float* b_d2  = (const float*)d_in[28];
  float* dout = (float*)d_out;

  // ---- workspace carve-up ----
  char* W8 = (char*)d_ws;
  size_t off = 0;
  auto alloc = [&](size_t bytes) -> void* {
    void* p = W8 + off;
    off = (off + bytes + 255) & ~(size_t)255;
    return p;
  };
  const size_t nPix32 = (size_t)B_ * PH_ * PW_ * 32;     // bf16 elements
  bf16* xb  = (bf16*)alloc(nPix32 * 2);
  bf16* tb  = (bf16*)alloc(nPix32 * 2);
  bf16* rb  = (bf16*)alloc(nPix32 * 2);
  bf16* fb  = (bf16*)alloc(nPix32 * 2);
  bf16* a1b = (bf16*)alloc(nPix32 * 2);
  bf16* a1w = (bf16*)alloc(nPix32 * 2);
  bf16* Xb  = (bf16*)alloc((size_t)B_ * PH_ * PW_ * XCS * 2);
  bf16* hb  = (bf16*)alloc((size_t)B_ * PH_ * PW_ * HCS * 2 + 4096);
  const size_t zeroBytes = off;                          // zero all activations
  float* crw = (float*)alloc((size_t)B_ * ND_ * H_ * W_ * 4);

  auto wpBytes = [](int Cin, int Cout) {
    return (size_t)((Cout + 15) / 16) * 9 * ((Cin + 31) / 32) * 512 * 2;
  };
  bf16* wc0p = (bf16*)alloc(wpBytes(32, 32));
  bf16* wr1p = (bf16*)alloc(wpBytes(32, 32));
  bf16* wr2p = (bf16*)alloc(wpBytes(32, 32));
  bf16* wg0p = (bf16*)alloc(wpBytes(41, 128));
  bf16* wg1p = (bf16*)alloc(wpBytes(169, 96));
  bf16* wg2p = (bf16*)alloc(wpBytes(265, 64));
  bf16* wg3p = (bf16*)alloc(wpBytes(329, 32));
  bf16* wd1p = (bf16*)alloc(wpBytes(361, 41));
  float* part = (float*)alloc((size_t)48 * STATP * 2 * 4);
  float* ss0  = (float*)alloc(48 * 2 * 4);
  float* ss1  = (float*)alloc(48 * 2 * 4);
  float* ssd  = (float*)alloc(48 * 2 * 4);

  const long HWl = (long)H_ * W_;
  const long Nbhw = (long)B_ * HWl;
  const float invN = 1.f / (float)Nbhw;

  // interior-origin helpers
  bf16* fbI  = fb  + ROW32 + 32;
  bf16* a1I  = a1b + ROW32 + 32;
  bf16* a1wI = a1w + ROW32 + 32;
  bf16* XI   = Xb  + ROWX + XCS;
  bf16* hbI  = hb  + ROWH + HCS;

  auto launchN = [&](long n) -> unsigned { return (unsigned)((n + 255) / 256); };

  auto pack = [&](const float* w, bf16* wp, int Cin, int Cout) {
    int cChunks = (Cin + 31) / 32;
    int nTiles  = (Cout + 15) / 16;
    long total  = (long)nTiles * 9 * cChunks * 512;
    pack_w<<<launchN(total), 256, 0, stream>>>(w, wp, Cin, Cout, cChunks, total);
  };

  auto conv = [&](const bf16* in, int inCS, long inRow, long inImg, int Cin,
                  const bf16* wp, const float* bias,
                  bf16* out, int outCS, long outRow, long outImg, int Cout,
                  int mode, const bf16* res, int resCS, long resRow, long resImg) {
    conv3x3_wmma<<<dim3(W_ / 64, H_, B_), 128, 0, stream>>>(
        in, inCS, inRow, inImg, Cin, wp, bias,
        out, outCS, outRow, outImg, Cout, mode, res, resCS, resRow, resImg);
  };

  // 1) zero all activation buffers (padded borders must be 0 every call)
  zero_k<<<8192, 256, 0, stream>>>((float4*)d_ws, (long)(zeroBytes / 16));

  // 2) pack all conv weights into WMMA B layout (bf16)
  pack(w_c0, wc0p, 32, 32);
  pack(w_r1, wr1p, 32, 32);
  pack(w_r2, wr2p, 32, 32);
  pack(w_g[0], wg0p, 41, 128);
  pack(w_g[1], wg1p, 169, 96);
  pack(w_g[2], wg2p, 265, 64);
  pack(w_g[3], wg3p, 329, 32);
  pack(w_d1, wd1p, 361, 41);

  // 3) shared feature extractor on both images
  for (int im = 0; im < 2; ++im) {
    stats_nchw<<<dim3(32, STATP), 256, 0, stream>>>(fimg[im], 32, HWl, Nbhw, part);
    stats_fin<<<1, 64, 0, stream>>>(part, 32, invN, bn0_g, bn0_b, ss0);
    bn_nchw2nhwc<<<launchN(Nbhw * 32), 256, 0, stream>>>(fimg[im], ss0, xb, Nbhw * 32);

    conv(xb, 32, ROW32, IMG32, 32, wc0p, b_c0, tb, 32, ROW32, IMG32, 32, 1,
         tb, 32, ROW32, IMG32);                                  // t = relu(conv0)
    conv(tb, 32, ROW32, IMG32, 32, wr1p, b_r1, rb, 32, ROW32, IMG32, 32, 1,
         tb, 32, ROW32, IMG32);                                  // r = relu(conv1)
    conv(rb, 32, ROW32, IMG32, 32, wr2p, b_r2, fb, 32, ROW32, IMG32, 32, 3,
         tb, 32, ROW32, IMG32);                                  // f = relu(conv2 + t)

    stats_nhwc<<<dim3(32, STATP), 256, 0, stream>>>(fbI, 32, ROW32, IMG32, part);
    stats_fin<<<1, 64, 0, stream>>>(part, 32, invN, bn1_g, bn1_b, ss1);
    if (im == 0)
      bn_nhwc<<<launchN(Nbhw * 32), 256, 0, stream>>>(fbI, 32, ROW32, IMG32,
          XI + CH_A0, XCS, ROWX, IMGX, ss1, 32, Nbhw * 32);      // a0 into X
    else
      bn_nhwc<<<launchN(Nbhw * 32), 256, 0, stream>>>(fbI, 32, ROW32, IMG32,
          a1I, 32, ROW32, IMG32, ss1, 32, Nbhw * 32);            // a1
  }

  // 4) warp a1 by disp0, correlation volume + box filter + 1x1 conv into X
  warp_k<<<launchN(Nbhw), 256, 0, stream>>>(a1I, disp0, a1wI, Nbhw);
  corr_raw_k<<<launchN(Nbhw), 256, 0, stream>>>(XI + CH_A0, a1wI, crw, Nbhw);
  corr_box_ca_k<<<launchN(Nbhw), 256, 0, stream>>>(crw, w_ca, b_ca, XI + CH_COST, Nbhw);

  // 5) dense regulator (each layer reads a contiguous channel tail of X)
  conv(Xb + CH_COST, XCS, ROWX, IMGX, 41,  wg0p, b_g[0], Xb + CH_G0, XCS, ROWX, IMGX, 128, 2, Xb, XCS, ROWX, IMGX);
  conv(Xb + CH_G0,   XCS, ROWX, IMGX, 169, wg1p, b_g[1], Xb + CH_G1, XCS, ROWX, IMGX, 96,  2, Xb, XCS, ROWX, IMGX);
  conv(Xb + CH_G1,   XCS, ROWX, IMGX, 265, wg2p, b_g[2], Xb + CH_G2, XCS, ROWX, IMGX, 64,  2, Xb, XCS, ROWX, IMGX);
  conv(Xb + CH_G2,   XCS, ROWX, IMGX, 329, wg3p, b_g[3], Xb + CH_G3, XCS, ROWX, IMGX, 32,  2, Xb, XCS, ROWX, IMGX);

  // 6) w_d1 (361 -> 41, lrelu), batchnorm
  conv(Xb, XCS, ROWX, IMGX, 361, wd1p, b_d1, hb, HCS, ROWH, IMGH, 41, 2, Xb, XCS, ROWX, IMGX);
  stats_nhwc<<<dim3(41, STATP), 256, 0, stream>>>(hbI, HCS, ROWH, IMGH, part);
  stats_fin<<<1, 64, 0, stream>>>(part, 41, invN, bnd_g, bnd_b, ssd);
  bn_nhwc<<<launchN(Nbhw * 41), 256, 0, stream>>>(hbI, HCS, ROWH, IMGH,
      hbI, HCS, ROWH, IMGH, ssd, 41, Nbhw * 41);

  // 7) head: tanh(conv 41->1)*MAXD + disp0 ; then emit x in NCHW fp32
  head_k<<<launchN(Nbhw), 256, 0, stream>>>(hb, w_d2, b_d2, disp0, dout, Nbhw);
  emit_x_k<<<launchN(Nbhw * 361), 256, 0, stream>>>(Xb, dout + Nbhw, Nbhw * 361);
}
